// DVAEncoder_PYG_69707319214191
// MI455X (gfx1250) — matrix operand order repaired and checked by reading, not compiled, not run on hardware
//
#include <hip/hip_runtime.h>
#include <hip/hip_bf16.h>

// ---------------------------------------------------------------------------
// D-VAE GRU encoder for MI455X (gfx1250, wave32, WMMA).
// Sequential scan over 2*128 node steps; per-step GEMMs done with
// v_wmma_f32_16x16x32_f16 (f16 in, f32 accumulate), recurrent state in f32.
// Message epilogue is fully branchless (mask folded into the multiply,
// edge-type one-hot bias resolved with selects over the 3 possible values).
// ---------------------------------------------------------------------------

#define B_   16
#define N_   128
#define HS_  128
#define NVT_ 4
#define NET_ 3
#define VS_  131   // HS + NET
#define L_   32
#define EPS_ 1e-5f

typedef _Float16 half_t;
typedef __attribute__((ext_vector_type(16))) _Float16 v16h;
typedef __attribute__((ext_vector_type(8)))  _Float16 v8h;
typedef __attribute__((ext_vector_type(8)))  float    v8f;

__device__ __forceinline__ float sigmoidf_(float x) {
  return 1.0f / (1.0f + __expf(-x));
}

__device__ __forceinline__ v8f wmma_f16(v16h a, v16h b, v8f c) {
  // D = A(16x32 f16) * B(32x16 f16) + C(16x16 f32)
  return __builtin_amdgcn_wmma_f32_16x16x32_f16(
      /*neg_a=*/false, a, /*neg_b=*/false, b,
      /*c_mod=*/(short)0, c, /*reuse_a=*/false, /*reuse_b=*/false);
}

// A-matrix fragment (16x32 f16, ISA 7.12.2): lane%16 = row M.
// lanes 0-15 hold K {kt*32+0..7, kt*32+16..23}; lanes 16-31 hold +8 / +24.
__device__ __forceinline__ v16h load_a_frag(const half_t* row, int ktile, int lane) {
  int hi = (lane & 16) ? 8 : 0;
  union { v16h v; v8h h[2]; } u;
  u.h[0] = *(const v8h*)(row + ktile * 32 + hi);
  u.h[1] = *(const v8h*)(row + ktile * 32 + hi + 16);
  return u.v;
}

// B-matrix fragment (32x16 f16): W is row-major (out x K=HS). lane%16 = col N
// (= output row of W); lanes 0-15 hold K {kt*32+0..7, kt*32+16..23}; lanes 16-31 hold +8 / +24.
__device__ __forceinline__ v16h load_b_frag(const half_t* W, int col, int ktile, int lane) {
  const half_t* p = W + (size_t)col * HS_ + ktile * 32 + ((lane & 16) ? 8 : 0);
  union { v16h v; v8h h[2]; } u;
  u.h[0] = *(const v8h*)(p);
  u.h[1] = *(const v8h*)(p + 16);
  return u.v;
}

// ---------------------------------------------------------------------------
// Prep: convert weights to f16, zero recurrent-state buffers (every call,
// since the harness does not re-initialize workspace between replays).
// ---------------------------------------------------------------------------
__global__ void prep_kernel(const float* Wg_f, const float* Wm_f,
                            const float* Wg_b, const float* Wm_b,
                            const float* Whh_f, const float* Whh_b,
                            half_t* Wg16f, half_t* Wm16f,
                            half_t* Wg16b, half_t* Wm16b,
                            half_t* Whh16f, half_t* Whh16b,
                            float* HfF, float* HbF,
                            half_t* Hf16, half_t* Hb16,
                            float* zerosF, half_t* zeros16) {
  int tid = blockIdx.x * blockDim.x + threadIdx.x;
  int stride = gridDim.x * blockDim.x;
  for (int i = tid; i < HS_ * HS_; i += stride) {
    int h = i >> 7, k = i & 127;                  // keep only first HS of VS cols
    Wg16f[i] = (half_t)Wg_f[h * VS_ + k];
    Wm16f[i] = (half_t)Wm_f[h * VS_ + k];
    Wg16b[i] = (half_t)Wg_b[h * VS_ + k];
    Wm16b[i] = (half_t)Wm_b[h * VS_ + k];
  }
  for (int i = tid; i < 3 * HS_ * HS_; i += stride) {
    Whh16f[i] = (half_t)Whh_f[i];
    Whh16b[i] = (half_t)Whh_b[i];
  }
  for (int i = tid; i < B_ * N_ * HS_; i += stride) {
    HfF[i] = 0.f; HbF[i] = 0.f;
    Hf16[i] = (half_t)0.f; Hb16[i] = (half_t)0.f;
  }
  for (int i = tid; i < B_ * HS_; i += stride) {
    zerosF[i] = 0.f; zeros16[i] = (half_t)0.f;
  }
}

// ---------------------------------------------------------------------------
// Gated message + masked reduce for one scan step:
//   msg[b,n,:] = sigmoid(Hall[b,n]@Wg[:, :HS].T + Wg[:,HS+et] + bg)
//              * (Hall[b,n]@Wm[:, :HS].T + Wm[:,HS+et])
//   H[b,:]    = sum_n mask[b,n] * msg[b,n,:]
// One workgroup per batch element; 8 waves = 8 M-tiles of n; K = HS = 4 tiles.
// ---------------------------------------------------------------------------
__global__ void msg_step_kernel(const half_t* __restrict__ Hall16,
                                const half_t* __restrict__ Wg16,
                                const half_t* __restrict__ Wm16,
                                const float*  __restrict__ Wg,
                                const float*  __restrict__ Wm,
                                const float*  __restrict__ bg,
                                const float*  __restrict__ adj,
                                const int*    __restrict__ etype,
                                int v, int sn, int so,
                                float* __restrict__ Hstep,
                                half_t* __restrict__ Hstep16) {
  __shared__ float sH[HS_];
  int tid = threadIdx.x;
  if (tid < HS_) sH[tid] = 0.f;
  __syncthreads();

  int b    = blockIdx.x;
  int lane = tid & 31;
  int wave = tid >> 5;
  int m    = lane & 15;
  int n0   = wave * 16;

  // A fragments: rows n0..n0+15 of Hall[b], all 4 k-tiles (reused 16x below)
  const half_t* row = Hall16 + ((size_t)(b * N_ + n0 + m)) * HS_;
  v16h a[4];
#pragma unroll
  for (int kt = 0; kt < 4; ++kt) a[kt] = load_a_frag(row, kt, lane);

  // mask + edge-type for the 8 rows this lane-half owns in the accumulator.
  // Unconditional loads: etype is defined for every (b,n,v); a zero mask
  // contributes exactly 0, so masking is folded into the multiply (no
  // divergent EXEC juggling in the hot loop).
  int rb = n0 + ((lane & 16) ? 8 : 0);
  const int baseIdx = b * N_ * N_ + so * v;
  float mks[8]; int ets[8];
#pragma unroll
  for (int r = 0; r < 8; ++r) {
    int idx = baseIdx + sn * (rb + r);
    mks[r] = adj[idx];
    ets[r] = etype[idx];
  }

  for (int nt = 0; nt < 8; ++nt) {
    v8f ag = {}, am = {};
#pragma unroll
    for (int kt = 0; kt < 4; ++kt) {
      ag = wmma_f16(a[kt], load_b_frag(Wg16, nt * 16 + m, kt, lane), ag);
      am = wmma_f16(a[kt], load_b_frag(Wm16, nt * 16 + m, kt, lane), am);
    }
    int col = nt * 16 + m;
    float bgc = bg[col];
    // NET=3 possible one-hot edge biases for this output column: preload all
    // three, resolve per row with selects (no gathers, no branches).
    const float* wgE = Wg + (size_t)col * VS_ + HS_;
    const float* wmE = Wm + (size_t)col * VS_ + HS_;
    float ge0 = wgE[0], ge1 = wgE[1], ge2 = wgE[2];
    float me0 = wmE[0], me1 = wmE[1], me2 = wmE[2];
    float partial = 0.f;
#pragma unroll
    for (int r = 0; r < 8; ++r) {
      int et = ets[r];
      float ge = (et == 0) ? ge0 : ((et == 1) ? ge1 : ge2);
      float me = (et == 0) ? me0 : ((et == 1) ? me1 : me2);
      float g  = ag[r] + bgc + ge;
      float mm = am[r] + me;
      partial += mks[r] * sigmoidf_(g) * mm;
    }
    atomicAdd(&sH[col], partial);
  }
  __syncthreads();
  if (tid < HS_) {
    float hv = sH[tid];
    Hstep[b * HS_ + tid]   = hv;
    Hstep16[b * HS_ + tid] = (half_t)hv;
  }
}

// ---------------------------------------------------------------------------
// GRU cell for one step:  h_v = GRU(onehot(node_type), H_in)
// gh = H_in @ Whh.T (M=16 batch, K=128, N=384) via WMMA; gi is a row-select
// of Wih (x is one-hot).  Wave w owns output cols 16w..16w+15 for all three
// gates (N-tiles w, w+8, w+16), so r/z/n combine in-register.
// ---------------------------------------------------------------------------
__global__ void gru_step_kernel(const half_t* __restrict__ Hin16, int strideH16,
                                const float*  __restrict__ HinF,  int strideHF,
                                const float*  __restrict__ Wih,
                                const half_t* __restrict__ Whh16,
                                const float*  __restrict__ bih,
                                const float*  __restrict__ bhh,
                                const int*    __restrict__ node_type, int v,
                                float* __restrict__ HallF,
                                half_t* __restrict__ Hall16) {
  __shared__ int snt[B_];
  int tid = threadIdx.x;
  if (tid < B_) snt[tid] = node_type[tid * N_ + v];
  __syncthreads();

  int lane = tid & 31;
  int wave = tid >> 5;
  int m    = lane & 15;

  const half_t* row = Hin16 + (size_t)m * strideH16;   // A row = batch m
  v16h a[4];
#pragma unroll
  for (int kt = 0; kt < 4; ++kt) a[kt] = load_a_frag(row, kt, lane);

  v8f accr = {}, accz = {}, accn = {};
#pragma unroll
  for (int kt = 0; kt < 4; ++kt) {
    accr = wmma_f16(a[kt], load_b_frag(Whh16, (wave)      * 16 + m, kt, lane), accr);
    accz = wmma_f16(a[kt], load_b_frag(Whh16, (wave + 8)  * 16 + m, kt, lane), accz);
    accn = wmma_f16(a[kt], load_b_frag(Whh16, (wave + 16) * 16 + m, kt, lane), accn);
  }

  int col = wave * 16 + m;                 // output hidden index, 0..127
  float bihr = bih[col], bihz = bih[HS_ + col], bihn = bih[2 * HS_ + col];
  float bhhr = bhh[col], bhhz = bhh[HS_ + col], bhhn = bhh[2 * HS_ + col];
  const float* WihR = Wih + (size_t)(col) * NVT_;
  const float* WihZ = Wih + (size_t)(HS_ + col) * NVT_;
  const float* WihN = Wih + (size_t)(2 * HS_ + col) * NVT_;

  int rb = (lane & 16) ? 8 : 0;
#pragma unroll
  for (int r = 0; r < 8; ++r) {
    int b = rb + r;
    int t = snt[b];
    float rr = sigmoidf_(WihR[t] + bihr + accr[r] + bhhr);
    float zz = sigmoidf_(WihZ[t] + bihz + accz[r] + bhhz);
    float hn = accn[r] + bhhn;
    float nn = tanhf(WihN[t] + bihn + rr * hn);
    float hprev = HinF[(size_t)b * strideHF + col];
    float h = (1.f - zz) * nn + zz * hprev;
    size_t o = ((size_t)b * N_ + v) * HS_ + col;
    HallF[o]  = h;
    Hall16[o] = (half_t)h;
  }
}

// ---------------------------------------------------------------------------
// Global head: Hg = BN(cat(Hf[:,N-1],Hb[:,0]) @ hgW.T + hgb); t = BN(relu(
// Hg @ gcW1.T + gcb1)); logit = t @ gcW2.T + gcb2.  All in one WG (tiny).
// ---------------------------------------------------------------------------
__global__ void global_head_kernel(const float* HfF, const float* HbF,
                                   const float* hgW, const float* hgb,
                                   const float* hgg, const float* hgbeta,
                                   const float* gcW1, const float* gcb1,
                                   const float* gcg, const float* gcbeta,
                                   const float* gcW2, const float* gcb2,
                                   float* out) {
  __shared__ float sIn[B_][2 * HS_];
  __shared__ float sHg[B_][HS_];
  __shared__ float sT[B_][2 * HS_];
  int tid = threadIdx.x;
  for (int i = tid; i < B_ * 2 * HS_; i += blockDim.x) {
    int b = i >> 8, k = i & 255;
    sIn[b][k] = (k < HS_) ? HfF[((size_t)b * N_ + (N_ - 1)) * HS_ + k]
                          : HbF[((size_t)b * N_) * HS_ + (k - HS_)];
  }
  __syncthreads();
  for (int i = tid; i < B_ * HS_; i += blockDim.x) {
    int b = i >> 7, j = i & 127;
    float s = hgb[j];
    for (int k = 0; k < 2 * HS_; ++k) s += sIn[b][k] * hgW[(size_t)j * 2 * HS_ + k];
    sHg[b][j] = s;
  }
  __syncthreads();
  if (tid < HS_) {
    float m = 0.f, m2 = 0.f;
    for (int b = 0; b < B_; ++b) { float x = sHg[b][tid]; m += x; m2 += x * x; }
    m /= B_;
    float var = m2 / B_ - m * m;
    float is = rsqrtf(var + EPS_);
    for (int b = 0; b < B_; ++b)
      sHg[b][tid] = hgg[tid] * (sHg[b][tid] - m) * is + hgbeta[tid];
  }
  __syncthreads();
  for (int i = tid; i < B_ * 2 * HS_; i += blockDim.x) {
    int b = i >> 8, c = i & 255;
    float s = gcb1[c];
    for (int j = 0; j < HS_; ++j) s += sHg[b][j] * gcW1[(size_t)c * HS_ + j];
    sT[b][c] = fmaxf(s, 0.f);
  }
  __syncthreads();
  if (tid < 2 * HS_) {
    float m = 0.f, m2 = 0.f;
    for (int b = 0; b < B_; ++b) { float x = sT[b][tid]; m += x; m2 += x * x; }
    m /= B_;
    float var = m2 / B_ - m * m;
    float is = rsqrtf(var + EPS_);
    for (int b = 0; b < B_; ++b)
      sT[b][tid] = gcg[tid] * (sT[b][tid] - m) * is + gcbeta[tid];
  }
  __syncthreads();
  if (tid < B_) {
    float s = gcb2[0];
    for (int c = 0; c < 2 * HS_; ++c) s += sT[tid][c] * gcW2[c];
    out[tid] = s;
  }
}

// --------------------------- Literal head ----------------------------------
__global__ void lit1_kernel(const float* HbF, const float* W1, const float* b1,
                            float* Z) {
  int idx0 = blockIdx.x * blockDim.x + threadIdx.x;
  int stride = gridDim.x * blockDim.x;
  for (int idx = idx0; idx < (B_ * L_) * (2 * HS_); idx += stride) {
    int i = idx >> 8;            // row in [0, 512)
    int c = idx & 255;
    int b = i >> 5;              // i / L
    int l = i & (L_ - 1);
    const float* h = HbF + ((size_t)b * N_ + 1 + l) * HS_;
    float s = b1[c];
    for (int k = 0; k < HS_; ++k) s += h[k] * W1[(size_t)c * HS_ + k];
    Z[(size_t)i * (2 * HS_) + c] = fmaxf(s, 0.f);
  }
}

__global__ void lit2_kernel(const float* Z, float* mv) {
  int c = threadIdx.x;           // 256 threads: one column each
  float m = 0.f, m2 = 0.f;
  for (int i = 0; i < B_ * L_; ++i) {
    float x = Z[(size_t)i * (2 * HS_) + c];
    m += x; m2 += x * x;
  }
  m /= (float)(B_ * L_);
  float var = m2 / (float)(B_ * L_) - m * m;
  mv[c] = m;
  mv[2 * HS_ + c] = var;
}

__global__ void lit3_kernel(const float* Z, const float* mv,
                            const float* g, const float* beta,
                            const float* W2, const float* b2,
                            const float* sol, float* out) {
  int i = blockIdx.x * blockDim.x + threadIdx.x;
  if (i < B_ * L_) {
    float s = b2[0];
    for (int c = 0; c < 2 * HS_; ++c) {
      float x = (Z[(size_t)i * (2 * HS_) + c] - mv[c]) * rsqrtf(mv[2 * HS_ + c] + EPS_);
      s += (g[c] * x + beta[c]) * W2[c];
    }
    out[B_ + i] = s;                       // predicted
    out[B_ + B_ * L_ + i] = sol[i];        // solutions passthrough
  }
}

// ---------------------------------------------------------------------------
extern "C" void kernel_launch(void* const* d_in, const int* in_sizes, int n_in,
                              void* d_out, int out_size, void* d_ws, size_t ws_size,
                              hipStream_t stream) {
  (void)in_sizes; (void)n_in; (void)out_size; (void)ws_size;

  const int*   node_type = (const int*)  d_in[0];
  const float* adj       = (const float*)d_in[1];
  const int*   etype     = (const int*)  d_in[2];
  const float* solutions = (const float*)d_in[3];
  const float* Wih_f = (const float*)d_in[4];
  const float* Whh_f = (const float*)d_in[5];
  const float* bih_f = (const float*)d_in[6];
  const float* bhh_f = (const float*)d_in[7];
  const float* Wih_b = (const float*)d_in[8];
  const float* Whh_b = (const float*)d_in[9];
  const float* bih_b = (const float*)d_in[10];
  const float* bhh_b = (const float*)d_in[11];
  const float* Wg_f  = (const float*)d_in[12];
  const float* bg_f  = (const float*)d_in[13];
  const float* Wm_f  = (const float*)d_in[14];
  const float* Wg_b  = (const float*)d_in[15];
  const float* bg_b  = (const float*)d_in[16];
  const float* Wm_b  = (const float*)d_in[17];
  const float* hg_W  = (const float*)d_in[18];
  const float* hg_b  = (const float*)d_in[19];
  const float* hg_g  = (const float*)d_in[20];
  const float* hg_be = (const float*)d_in[21];
  const float* gc_W1 = (const float*)d_in[22];
  const float* gc_b1 = (const float*)d_in[23];
  const float* gc_g  = (const float*)d_in[24];
  const float* gc_be = (const float*)d_in[25];
  const float* gc_W2 = (const float*)d_in[26];
  const float* gc_b2 = (const float*)d_in[27];
  const float* lc_W1 = (const float*)d_in[28];
  const float* lc_b1 = (const float*)d_in[29];
  const float* lc_g  = (const float*)d_in[30];
  const float* lc_be = (const float*)d_in[31];
  const float* lc_W2 = (const float*)d_in[32];
  const float* lc_b2 = (const float*)d_in[33];

  // ---- workspace partition (all 256B aligned) -----------------------------
  char* ws = (char*)d_ws;
  auto take = [&](size_t bytes) -> void* {
    void* p = (void*)ws;
    ws += (bytes + 255) & ~(size_t)255;
    return p;
  };
  float*  HfF     = (float*) take((size_t)B_ * N_ * HS_ * 4);
  float*  HbF     = (float*) take((size_t)B_ * N_ * HS_ * 4);
  half_t* Hf16    = (half_t*)take((size_t)B_ * N_ * HS_ * 2);
  half_t* Hb16    = (half_t*)take((size_t)B_ * N_ * HS_ * 2);
  half_t* Wg16f   = (half_t*)take((size_t)HS_ * HS_ * 2);
  half_t* Wm16f   = (half_t*)take((size_t)HS_ * HS_ * 2);
  half_t* Wg16b   = (half_t*)take((size_t)HS_ * HS_ * 2);
  half_t* Wm16b   = (half_t*)take((size_t)HS_ * HS_ * 2);
  half_t* Whh16f  = (half_t*)take((size_t)3 * HS_ * HS_ * 2);
  half_t* Whh16b  = (half_t*)take((size_t)3 * HS_ * HS_ * 2);
  float*  zerosF  = (float*) take((size_t)B_ * HS_ * 4);
  half_t* zeros16 = (half_t*)take((size_t)B_ * HS_ * 2);
  float*  Hstep   = (float*) take((size_t)B_ * HS_ * 4);
  half_t* Hstep16 = (half_t*)take((size_t)B_ * HS_ * 2);
  float*  Zlit    = (float*) take((size_t)(B_ * L_) * (2 * HS_) * 4);
  float*  mv      = (float*) take((size_t)2 * (2 * HS_) * 4);

  float* out = (float*)d_out;

  // ---- prep: f16 weight copies + zero state -------------------------------
  prep_kernel<<<256, 256, 0, stream>>>(Wg_f, Wm_f, Wg_b, Wm_b, Whh_f, Whh_b,
                                       Wg16f, Wm16f, Wg16b, Wm16b,
                                       Whh16f, Whh16b,
                                       HfF, HbF, Hf16, Hb16, zerosF, zeros16);

  // ---- forward scan: order 0..127, mask = adj[:, n, v] (sn=N, so=1) -------
  gru_step_kernel<<<1, 256, 0, stream>>>(zeros16, HS_, zerosF, HS_,
                                         Wih_f, Whh16f, bih_f, bhh_f,
                                         node_type, 0, HfF, Hf16);
  for (int v = 1; v < N_; ++v) {
    msg_step_kernel<<<B_, 256, 0, stream>>>(Hf16, Wg16f, Wm16f, Wg_f, Wm_f,
                                            bg_f, adj, etype, v,
                                            /*sn=*/N_, /*so=*/1,
                                            Hstep, Hstep16);
    gru_step_kernel<<<1, 256, 0, stream>>>(Hstep16, HS_, Hstep, HS_,
                                           Wih_f, Whh16f, bih_f, bhh_f,
                                           node_type, v, HfF, Hf16);
  }

  // ---- backward scan: order 127..0, H0 = Hf[:, N-1], mask = adj[:, v, n] --
  gru_step_kernel<<<1, 256, 0, stream>>>(Hf16 + (size_t)(N_ - 1) * HS_, N_ * HS_,
                                         HfF + (size_t)(N_ - 1) * HS_, N_ * HS_,
                                         Wih_b, Whh16b, bih_b, bhh_b,
                                         node_type, N_ - 1, HbF, Hb16);
  for (int v = N_ - 2; v >= 0; --v) {
    msg_step_kernel<<<B_, 256, 0, stream>>>(Hb16, Wg16b, Wm16b, Wg_b, Wm_b,
                                            bg_b, adj, etype, v,
                                            /*sn=*/1, /*so=*/N_,
                                            Hstep, Hstep16);
    gru_step_kernel<<<1, 256, 0, stream>>>(Hstep16, HS_, Hstep, HS_,
                                           Wih_b, Whh16b, bih_b, bhh_b,
                                           node_type, v, HbF, Hb16);
  }

  // ---- heads --------------------------------------------------------------
  global_head_kernel<<<1, 256, 0, stream>>>(HfF, HbF, hg_W, hg_b, hg_g, hg_be,
                                            gc_W1, gc_b1, gc_g, gc_be,
                                            gc_W2, gc_b2, out);
  lit1_kernel<<<32, 256, 0, stream>>>(HbF, lc_W1, lc_b1, Zlit);
  lit2_kernel<<<1, 256, 0, stream>>>(Zlit, mv);
  lit3_kernel<<<2, 256, 0, stream>>>(Zlit, mv, lc_g, lc_be, lc_W2, lc_b2,
                                     solutions, out);
}
